// Attention_87428354277788
// MI455X (gfx1250) — matrix operand order repaired and checked by reading, not compiled, not run on hardware
//
#include <hip/hip_runtime.h>

typedef __bf16 bf16;
typedef __bf16 v16bf __attribute__((ext_vector_type(16)));
typedef __bf16 v4bf  __attribute__((ext_vector_type(4)));
typedef float  v8f   __attribute__((ext_vector_type(8)));
typedef float  f4    __attribute__((ext_vector_type(4)));

union Frag { v16bf v; f4 h[2]; };

__device__ __forceinline__ v8f wmma_bf16(const Frag& a, const Frag& b, v8f c) {
  return __builtin_amdgcn_wmma_f32_16x16x32_bf16(false, a.v, false, b.v, (short)0, c, false, false);
}

// CDNA5 async copy: global -> LDS, 16 bytes per lane, tracked by ASYNCcnt.
// Low 32 bits of a generic pointer to __shared__ are the LDS byte address.
__device__ __forceinline__ void async_ld_b128(void* lds_dst, const void* gsrc) {
  unsigned loff = (unsigned)(size_t)lds_dst;
  asm volatile("global_load_async_to_lds_b128 %0, %1, off"
               :: "v"(loff), "v"(gsrc) : "memory");
}
__device__ __forceinline__ void wait_async0() {
  asm volatile("s_wait_asynccnt 0" ::: "memory");
}

// ---------------- conversion kernels ----------------
__global__ void cvt_kernel4(const float* __restrict__ src, bf16* __restrict__ dst, int n4) {
  int i = blockIdx.x * blockDim.x + threadIdx.x;
  if (i < n4) {
    f4 v = *(const f4*)(src + (size_t)i * 4);
    v4bf o; o[0] = (bf16)v[0]; o[1] = (bf16)v[1]; o[2] = (bf16)v[2]; o[3] = (bf16)v[3];
    *(v4bf*)(dst + (size_t)i * 4) = o;
  }
}

// W [R][C] row-major -> WT [C][R] row-major (i.e. [N][K] for GEMM B)
__global__ void cvt_T_kernel(const float* __restrict__ W, bf16* __restrict__ WT, int R, int C) {
  int i = blockIdx.x * blockDim.x + threadIdx.x;
  if (i < R * C) {
    int r = i / C, c = i % C;
    WT[(size_t)c * R + r] = (bf16)W[i];
  }
}

// ---------------- shared GEMM core: 64x128 tile, BK=64, double-buffered async LDS ----------------
#define AST 72   // LDS row stride in halfs (64 + 8 pad, keeps 16B alignment, kills bank conflicts)

template<int KK>
__device__ __forceinline__ void gemm_core(
    const bf16* __restrict__ A, const bf16* __restrict__ BT,
    int m0, int n0, int tid,
    bf16 (&Als)[2][64 * AST], bf16 (&Bls)[2][128 * AST],
    v8f (&acc)[2][2])
{
  const int lane = tid & 31, w = tid >> 5;
  const int wm = w & 1, wn = w >> 1;
  const int l16 = lane & 15, ahi = lane >> 4;
  const int akoff = ahi * 16, bkoff = ahi * 16;

  // staging assignment: A tile 64x64 halfs (16/thread), B tile 128x64 halfs (32/thread)
  const int ar = tid >> 2, aq = tid & 3;
  const int br = tid >> 1, bq = tid & 1;

  const bf16* asrc = A  + (size_t)(m0 + ar) * KK + aq * 16;
  const bf16* bsrc = BT + (size_t)(n0 + br) * KK + bq * 32;
  const int aoff = ar * AST + aq * 16;
  const int boff = br * AST + bq * 32;

  // prologue: stage K-block 0 into buffer 0
  async_ld_b128(&Als[0][aoff],      asrc);
  async_ld_b128(&Als[0][aoff + 8],  asrc + 8);
  async_ld_b128(&Bls[0][boff],      bsrc);
  async_ld_b128(&Bls[0][boff + 8],  bsrc + 8);
  async_ld_b128(&Bls[0][boff + 16], bsrc + 16);
  async_ld_b128(&Bls[0][boff + 24], bsrc + 24);
  wait_async0();
  __syncthreads();

  constexpr int NK = KK >> 6;   // K / 64, compile-time => full unroll, constant buffer indices
#pragma unroll
  for (int i = 0; i < NK; ++i) {
    if (i + 1 < NK) {
      const int nb = (i + 1) & 1;
      const int kb = (i + 1) << 6;
      async_ld_b128(&Als[nb][aoff],      asrc + kb);
      async_ld_b128(&Als[nb][aoff + 8],  asrc + kb + 8);
      async_ld_b128(&Bls[nb][boff],      bsrc + kb);
      async_ld_b128(&Bls[nb][boff + 8],  bsrc + kb + 8);
      async_ld_b128(&Bls[nb][boff + 16], bsrc + kb + 16);
      async_ld_b128(&Bls[nb][boff + 24], bsrc + kb + 24);
    }
    const int cb = i & 1;
#pragma unroll
    for (int kc = 0; kc < 2; ++kc) {
      Frag a[2], b[2];
#pragma unroll
      for (int rt = 0; rt < 2; ++rt) {
        const bf16* ap = &Als[cb][(wm * 32 + rt * 16 + l16) * AST + kc * 32 + akoff];
        a[rt].h[0] = *(const f4*)ap;
        a[rt].h[1] = *(const f4*)(ap + 8);
      }
#pragma unroll
      for (int ct = 0; ct < 2; ++ct) {
        const bf16* bp = &Bls[cb][(wn * 32 + ct * 16 + l16) * AST + kc * 32 + bkoff];
        b[ct].h[0] = ((const f4*)bp)[0];
        b[ct].h[1] = ((const f4*)bp)[1];
      }
#pragma unroll
      for (int rt = 0; rt < 2; ++rt)
#pragma unroll
        for (int ct = 0; ct < 2; ++ct)
          acc[rt][ct] = wmma_bf16(a[rt], b[ct], acc[rt][ct]);
    }
    wait_async0();     // own async loads into next buffer landed
    __syncthreads();   // everyone done (frag loads were consumed by wmma above)
  }
}

// ---------------- QKV GEMM with scatter epilogue ----------------
__global__ __launch_bounds__(256) void gemm_qkv_kernel(
    const bf16* __restrict__ A, const bf16* __restrict__ BT,
    const float* __restrict__ b_qkv,
    bf16* __restrict__ q, bf16* __restrict__ k, bf16* __restrict__ vT)
{
  __shared__ __align__(16) bf16 Als[2][64 * AST];
  __shared__ __align__(16) bf16 Bls[2][128 * AST];
  const int tid = threadIdx.x;
  const int lane = tid & 31, w = tid >> 5;
  const int wm = w & 1, wn = w >> 1;
  const int m0 = blockIdx.y * 64, n0 = blockIdx.x * 128;
  const int l16 = lane & 15, ahi = lane >> 4;

  v8f acc[2][2] = { { {}, {} }, { {}, {} } };
  gemm_core<768>(A, BT, m0, n0, tid, Als, Bls, acc);

#pragma unroll
  for (int rt = 0; rt < 2; ++rt)
#pragma unroll
    for (int ct = 0; ct < 2; ++ct) {
      int col  = n0 + wn * 32 + ct * 16 + l16;
      int ksel = col / 768, rem = col % 768;
      int h = rem >> 6, d = rem & 63;
      float bias = b_qkv[col];
#pragma unroll
      for (int vr = 0; vr < 8; ++vr) {
        int row = m0 + wm * 32 + rt * 16 + vr + ahi * 8;
        int bb = row >> 8, n = row & 255;
        float val = acc[rt][ct][vr] + bias;
        size_t bh = (size_t)bb * 12 + h;
        if (ksel == 0)      q [(bh * 256 + n) * 64 + d] = (bf16)(val * 0.125f); // pre-scale q
        else if (ksel == 1) k [(bh * 256 + n) * 64 + d] = (bf16)val;
        else                vT[(bh * 64 + d) * 256 + n] = (bf16)val;            // transposed v
      }
    }
}

// ---------------- output projection GEMM, f32 epilogue ----------------
__global__ __launch_bounds__(256) void gemm_out_kernel(
    const bf16* __restrict__ A, const bf16* __restrict__ BT,
    const float* __restrict__ b_out, float* __restrict__ out)
{
  __shared__ __align__(16) bf16 Als[2][64 * AST];
  __shared__ __align__(16) bf16 Bls[2][128 * AST];
  const int tid = threadIdx.x;
  const int lane = tid & 31, w = tid >> 5;
  const int wm = w & 1, wn = w >> 1;
  const int m0 = blockIdx.y * 64, n0 = blockIdx.x * 128;
  const int l16 = lane & 15, ahi = lane >> 4;

  v8f acc[2][2] = { { {}, {} }, { {}, {} } };
  gemm_core<768>(A, BT, m0, n0, tid, Als, Bls, acc);

#pragma unroll
  for (int rt = 0; rt < 2; ++rt)
#pragma unroll
    for (int ct = 0; ct < 2; ++ct) {
      int col = n0 + wn * 32 + ct * 16 + l16;
      float bias = b_out[col];
#pragma unroll
      for (int vr = 0; vr < 8; ++vr) {
        int row = m0 + wm * 32 + rt * 16 + vr + ahi * 8;
        out[(size_t)row * 768 + col] = acc[rt][ct][vr] + bias;
      }
    }
}

// ---------------- attention: one block per (b,h) ----------------
__global__ __launch_bounds__(256) void attn_kernel(
    const bf16* __restrict__ q, const bf16* __restrict__ k, const bf16* __restrict__ vT,
    const int* __restrict__ rel_ids, const float* __restrict__ rel_table,
    bf16* __restrict__ attn_out)
{
  __shared__ __align__(16) bf16 k_lds[256 * 72];
  __shared__ __align__(16) bf16 v_lds[64 * 264];
  __shared__ __align__(16) bf16 p_lds[8 * 16 * 40];

  const int bh = blockIdx.x;
  const int b = bh / 12, h = bh % 12;
  const int tid = threadIdx.x, lane = tid & 31, w = tid >> 5;
  const int l16 = lane & 15, ahi = lane >> 4;

  const bf16* qg = q  + (size_t)bh * 256 * 64;
  const bf16* kg = k  + (size_t)bh * 256 * 64;
  const bf16* vg = vT + (size_t)bh * 64 * 256;

  { // stage K: one 64-elem row per thread, async path
    int r = tid;
    const bf16* src = kg + r * 64;
    bf16* dst = k_lds + r * 72;
#pragma unroll
    for (int j = 0; j < 8; ++j) async_ld_b128(dst + j * 8, src + j * 8);
  }
  { // stage V^T: 64 rows x 256, 4 threads/row, async path
    int r = tid >> 2, qd = tid & 3;
    const bf16* src = vg + r * 256 + qd * 64;
    bf16* dst = v_lds + r * 264 + qd * 64;
#pragma unroll
    for (int j = 0; j < 8; ++j) async_ld_b128(dst + j * 8, src + j * 8);
  }
  wait_async0();
  __syncthreads();

  bf16* pw = p_lds + w * 16 * 40;

  for (int it = 0; it < 2; ++it) {
    const int nt = w + it * 8;
    const int n0 = nt * 16;

    // Q A-fragments (K = 0..31 and 32..63), direct from global (128-bit loads)
    Frag a0, a1;
    const bf16* qrow = qg + (size_t)(n0 + l16) * 64;
    a0.h[0] = *(const f4*)(qrow + ahi * 16);
    a0.h[1] = *(const f4*)(qrow + 8 + ahi * 16);
    a1.h[0] = *(const f4*)(qrow + 32 + ahi * 16);
    a1.h[1] = *(const f4*)(qrow + 40 + ahi * 16);

    // scores: 16 column tiles of 16
    v8f s[16];
#pragma unroll
    for (int jt = 0; jt < 16; ++jt) {
      Frag b0, b1;
      const bf16* kp = k_lds + (jt * 16 + l16) * 72 + ahi * 16;
      b0.h[0] = ((const f4*)kp)[0];
      b0.h[1] = ((const f4*)kp)[1];
      b1.h[0] = ((const f4*)(kp + 32))[0];
      b1.h[1] = ((const f4*)(kp + 32))[1];
      v8f c = {};
      c = wmma_bf16(a0, b0, c);
      c = wmma_bf16(a1, b1, c);
      s[jt] = c;
    }

    // bias gather + softmax (C layout: lane = col l16, VGPR vr = row vr + 8*ahi)
#pragma unroll
    for (int vr = 0; vr < 8; ++vr) {
      int n = n0 + vr + ahi * 8;
      const int* idrow = rel_ids + (size_t)n * 256;
      float vals[16];
      float mx = -1e30f;
#pragma unroll
      for (int jt = 0; jt < 16; ++jt) {
        int m = jt * 16 + l16;
        float sv = s[jt][vr] + rel_table[(size_t)idrow[m] * 12 + h];
        vals[jt] = sv;
        mx = fmaxf(mx, sv);
      }
#pragma unroll
      for (int off = 1; off <= 8; off <<= 1)
        mx = fmaxf(mx, __shfl_xor(mx, off, 32));
      float sum = 0.f;
#pragma unroll
      for (int jt = 0; jt < 16; ++jt) {
        float e = __expf(vals[jt] - mx);
        vals[jt] = e;
        sum += e;
      }
#pragma unroll
      for (int off = 1; off <= 8; off <<= 1)
        sum += __shfl_xor(sum, off, 32);
      float inv = 1.f / sum;
#pragma unroll
      for (int jt = 0; jt < 16; ++jt)
        s[jt][vr] = vals[jt] * inv;
    }

    // P @ V: re-layout P (C-layout -> A-fragment) 32 columns at a time via tiny LDS bounce
    v8f o[4] = { {}, {}, {}, {} };
#pragma unroll
    for (int c = 0; c < 8; ++c) {
#pragma unroll
      for (int t = 0; t < 2; ++t) {
        int jt = c * 2 + t;
#pragma unroll
        for (int vr = 0; vr < 8; ++vr)
          pw[(vr + ahi * 8) * 40 + t * 16 + l16] = (bf16)s[jt][vr];
      }
      asm volatile("s_wait_dscnt 0" ::: "memory");
      Frag a;
      const bf16* pp = pw + l16 * 40 + ahi * 16;
      a.h[0] = *(const f4*)pp;
      a.h[1] = *(const f4*)(pp + 8);
#pragma unroll
      for (int dt = 0; dt < 4; ++dt) {
        Frag bb;
        const bf16* vp = v_lds + (dt * 16 + l16) * 264 + c * 32 + ahi * 16;
        bb.h[0] = ((const f4*)vp)[0];
        bb.h[1] = ((const f4*)vp)[1];
        o[dt] = wmma_bf16(a, bb, o[dt]);
      }
      asm volatile("s_wait_dscnt 0" ::: "memory");
    }

    // store to attn_out [b, n, h*64+d] as bf16
#pragma unroll
    for (int dt = 0; dt < 4; ++dt) {
      int col = h * 64 + dt * 16 + l16;
#pragma unroll
      for (int vr = 0; vr < 8; ++vr) {
        int n = n0 + vr + ahi * 8;
        attn_out[((size_t)b * 256 + n) * 768 + col] = (bf16)o[dt][vr];
      }
    }
  }
}

// ---------------- host launch ----------------
extern "C" void kernel_launch(void* const* d_in, const int* in_sizes, int n_in,
                              void* d_out, int out_size, void* d_ws, size_t ws_size,
                              hipStream_t stream)
{
  const float* x     = (const float*)d_in[0];   // [128,256,768]
  const float* Wqkv  = (const float*)d_in[1];   // [768,2304]
  const float* bqkv  = (const float*)d_in[2];   // [2304]
  const float* Wout  = (const float*)d_in[3];   // [768,768]
  const float* bout  = (const float*)d_in[4];   // [768]
  const float* table = (const float*)d_in[5];   // [961,12]
  const int*   ids   = (const int*)d_in[6];     // [65536]
  float* out = (float*)d_out;

  char* ws = (char*)d_ws;
  size_t off = 0;
  auto alloc = [&](size_t bytes) { void* p = ws + off; off += (bytes + 255) & ~(size_t)255; return p; };

  bf16* xb    = (bf16*)alloc(32768ull * 768 * 2);
  bf16* WqkvT = (bf16*)alloc(2304ull * 768 * 2);
  bf16* WoutT = (bf16*)alloc(768ull * 768 * 2);
  bf16* qb    = (bf16*)alloc(128ull * 12 * 256 * 64 * 2);
  bf16* kb    = (bf16*)alloc(128ull * 12 * 256 * 64 * 2);
  bf16* vTb   = (bf16*)alloc(128ull * 12 * 64 * 256 * 2);
  bf16* attn  = xb;   // alias: x_bf16 is dead after the QKV GEMM

  cvt_kernel4 <<<25165824 / 4 / 256, 256, 0, stream>>>(x, xb, 25165824 / 4);
  cvt_T_kernel<<<(768 * 2304 + 255) / 256, 256, 0, stream>>>(Wqkv, WqkvT, 768, 2304);
  cvt_T_kernel<<<(768 * 768 + 255) / 256, 256, 0, stream>>>(Wout, WoutT, 768, 768);

  gemm_qkv_kernel<<<dim3(2304 / 128, 32768 / 64), 256, 0, stream>>>(
      xb, WqkvT, bqkv, qb, kb, vTb);

  attn_kernel<<<128 * 12, 256, 0, stream>>>(qb, kb, vTb, ids, table, attn);

  gemm_out_kernel<<<dim3(768 / 128, 32768 / 64), 256, 0, stream>>>(
      attn, WoutT, bout, out);
}
